// DualEncoderEpsNetwork_9844065042515
// MI455X (gfx1250) — matrix operand order repaired and checked by reading, not compiled, not run on hardware
//
#include <hip/hip_runtime.h>
#include <hip/hip_bf16.h>

typedef __bf16 bf16_t;
typedef __attribute__((ext_vector_type(16))) __bf16 v16bf;
typedef __attribute__((ext_vector_type(8)))  __bf16 v8bf;
typedef __attribute__((ext_vector_type(8)))  float  v8f;
typedef __attribute__((ext_vector_type(4)))  float  v4f;

static constexpr int NE = 128 * 32 * 32;   // 131072 edges

// ---------------------------------------------------------------- WMMA core
__device__ __forceinline__ v8f wmma_bf(v16bf a, v16bf b, v8f c) {
  return __builtin_amdgcn_wmma_f32_16x16x32_bf16(false, a, false, b, (short)0, c,
                                                 false, false);
}

// A-matrix 16x32 bf16 tile (ISA 7.12.2): lane L -> M=L&15; per lane two
// contiguous 8-element K runs at k0+half*8 and k0+16+half*8.  f32 source.
__device__ __forceinline__ v16bf load_a(const float* base, int ld, int k0) {
  int lane = threadIdx.x & 31;
  const float* row = base + (size_t)(lane & 15) * ld + k0 + (lane >> 4) * 8;
  v4f x0 = ((const v4f*)row)[0];
  v4f x1 = ((const v4f*)row)[1];
  v4f x2 = ((const v4f*)row)[4];   // +16 floats
  v4f x3 = ((const v4f*)row)[5];
  v16bf a;
#pragma unroll
  for (int e = 0; e < 4; ++e) {
    a[e] = (bf16_t)x0[e];  a[4 + e] = (bf16_t)x1[e];
    a[8 + e] = (bf16_t)x2[e]; a[12 + e] = (bf16_t)x3[e];
  }
  return a;
}

// bf16 source (global attr rows): two 16-byte vector loads per lane
__device__ __forceinline__ v16bf load_a(const bf16_t* base, int ld, int k0) {
  int lane = threadIdx.x & 31;
  const bf16_t* row = base + (size_t)(lane & 15) * ld + k0 + (lane >> 4) * 8;
  v8bf lo = *(const v8bf*)row;
  v8bf hi = *(const v8bf*)(row + 16);
  return __builtin_shufflevector(lo, hi, 0, 1, 2, 3, 4, 5, 6, 7, 8, 9, 10, 11,
                                 12, 13, 14, 15);
}

// B-matrix 32x16 from TRANSPOSED LDS weights Wt[N][ldk] (bf16): lane L ->
// N=L&15, K runs k0+half*8 .. +8 and k0+16+half*8 .. +8 -> two b128 LDS loads.
__device__ __forceinline__ v16bf load_b_t(const bf16_t* Wt, int ldk, int k0,
                                          int n0) {
  int lane = threadIdx.x & 31;
  const bf16_t* col = Wt + (size_t)(n0 + (lane & 15)) * ldk + k0 + (lane >> 4) * 8;
  v8bf lo = *(const v8bf*)col;
  v8bf hi = *(const v8bf*)(col + 16);
  return __builtin_shufflevector(lo, hi, 0, 1, 2, 3, 4, 5, 6, 7, 8, 9, 10, 11,
                                 12, 13, 14, 15);
}

// B-matrix from global f32 W[K][ld] (only used by the small cond kernel)
__device__ __forceinline__ v16bf load_b_g(const float* W, int ld, int k0, int n0) {
  int lane = threadIdx.x & 31;
  const float* col = W + (size_t)(k0 + (lane >> 4) * 8) * ld + n0 + (lane & 15);
  v16bf b;
#pragma unroll
  for (int e = 0; e < 16; ++e) b[e] = (bf16_t)col[(size_t)(e < 8 ? e : e + 8) * ld];
  return b;
}

enum { ACT_NONE = 0, ACT_RELU = 1, ACT_GELU = 2, ACT_SSP = 3 };

__device__ __forceinline__ float apply_act(float x, int act) {
  switch (act) {
    case ACT_RELU: return fmaxf(x, 0.f);
    case ACT_GELU: {
      float x3 = x * x * x;
      float y = 1.5957691216f * (x + 0.044715f * x3);  // 2*0.79788456*(...)
      float t = 1.f - 2.f / (__expf(y) + 1.f);          // tanh via exp
      return 0.5f * x * (1.f + t);
    }
    case ACT_SSP: {
      float sp = (x > 15.f) ? x : __logf(1.f + __expf(x));
      return sp - 0.69314718056f;
    }
  }
  return x;
}

// stage global f32 W[K][N] into LDS as TRANSPOSED bf16 Wt[N][K+8] (row pad
// keeps 16B alignment and staggers LDS banks across n)
__device__ __forceinline__ void stage_w_t(bf16_t* dst, const float* src, int K,
                                          int N) {
  int kp = K + 8;
  for (int i = threadIdx.x; i < K * N; i += blockDim.x) {
    int k = i / N, n = i - k * N;  // coalesced global read
    dst[n * kp + k] = (bf16_t)src[i];
  }
}

// one 16-row M-tile: C = act(A @ W + bias); A: LDS f32 [16][lda], W: global
// f32 [K][ldw], C: LDS f32 [16][ldc]; wave w covers n-tiles w*16, w*16+128,...
__device__ void gemm_tile_g(const float* Als, int lda, const float* Wg, int ldw,
                            const float* bias, float* Cls, int ldc, int N, int K,
                            int act) {
  int wv = threadIdx.x >> 5, lane = threadIdx.x & 31;
  for (int nt = wv * 16; nt < N; nt += 128) {
    v8f acc = {};
    for (int k0 = 0; k0 < K; k0 += 32)
      acc = wmma_bf(load_a(Als, lda, k0), load_b_g(Wg, ldw, k0, nt), acc);
    int n = nt + (lane & 15), mb = (lane >> 4) * 8;
    float bv = bias[n];
#pragma unroll
    for (int r = 0; r < 8; ++r)
      Cls[(mb + r) * ldc + n] = apply_act(acc[r] + bv, act);
  }
}

// ---------------------------------------------------------------- k_graph
__global__ __launch_bounds__(1024) void k_graph(
    const float* pos, const int* bond, float* dist, int* etype, float* Cm,
    float* validf, float* localf, float* out_et, float* out_el, float* out_valid,
    float* out_local) {
  __shared__ unsigned char sA[32][32], sB2[32][32], sB3[32][32];
  __shared__ float sp[32][3];
  int b = blockIdx.x;
  int t = threadIdx.x, i = t >> 5, j = t & 31;
  if (t < 96) sp[t / 3][t % 3] = pos[(size_t)b * 96 + t];
  int bt = bond[(size_t)b * 1024 + t];
  sA[i][j] = bt > 0;
  __syncthreads();
  {
    unsigned char r = 0;
    for (int k = 0; k < 32; ++k) r |= (unsigned char)(sA[i][k] & sA[k][j]);
    sB2[i][j] = r;
  }
  __syncthreads();
  {
    unsigned char r = 0;
    for (int k = 0; k < 32; ++k) r |= (unsigned char)(sB2[i][k] & sA[k][j]);
    sB3[i][j] = r;
  }
  __syncthreads();
  bool Aij = sA[i][j] != 0, I = (i == j);
  bool h2 = sB2[i][j] && !Aij && !I;
  bool h3 = sB3[i][j] && !sB2[i][j] && !Aij && !I;
  int et = Aij ? bt : (h2 ? 5 : (h3 ? 6 : 0));
  float dx = sp[i][0] - sp[j][0], dy = sp[i][1] - sp[j][1], dz = sp[i][2] - sp[j][2];
  float d = sqrtf(dx * dx + dy * dy + dz * dz + 1e-12f);
  bool valid = ((et > 0) || (d < 10.0f)) && !I;
  bool local = et > 0;
  float c = ((d <= 10.0f) && valid) ? 1.f : 0.f;
  size_t e = (size_t)b * 1024 + t;
  dist[e] = d; etype[e] = et; Cm[e] = c;
  validf[e] = valid ? 1.f : 0.f; localf[e] = local ? 1.f : 0.f;
  out_et[e] = (float)et; out_el[e] = d;
  out_valid[e] = valid ? 1.f : 0.f; out_local[e] = local ? 1.f : 0.f;
}

// ---------------------------------------------------------------- k_cond
// padded LDS: sA 16x1028, sH 16x516, sCat 16x644
__global__ __launch_bounds__(256) void k_cond(
    const float* spectrum, const int* tstep, const float* Ws1, const float* bs1,
    const float* Ws2, const float* bs2, const float* Wt1, const float* bt1,
    const float* Wt2, const float* bt2, const float* Wc1, const float* bc1,
    const float* Wc2, const float* bc2, float* cond) {
  extern __shared__ char smc[];
  float* sA = (float*)smc;                  // 16*1028*4 = 65792
  float* sH = (float*)(smc + 65792);        // 16*516*4  = 33024
  float* sCat = (float*)(smc + 98816);      // 16*644*4  = 41216
  int b0 = blockIdx.x * 16;
  for (int idx = threadIdx.x; idx < 16 * 1024; idx += 256)
    sA[(idx >> 10) * 1028 + (idx & 1023)] =
        spectrum[(size_t)(b0 + (idx >> 10)) * 1024 + (idx & 1023)];
  __syncthreads();
  gemm_tile_g(sA, 1028, Ws1, 128, bs1, sH, 516, 128, 1024, ACT_GELU);
  __syncthreads();
  gemm_tile_g(sH, 516, Ws2, 128, bs2, sCat, 644, 128, 128, ACT_NONE);
  __syncthreads();
  for (int idx = threadIdx.x; idx < 16 * 128; idx += 256) {
    int r = idx >> 7, c = idx & 127;
    float t = (float)tstep[b0 + r];
    int hc = c & 63;
    float v = t * __expf(-(float)hc * 0.14619587f);  // log(1e4)/63
    sA[r * 1028 + c] = (c < 64) ? __sinf(v) : __cosf(v);
  }
  __syncthreads();
  gemm_tile_g(sA, 1028, Wt1, 512, bt1, sH, 516, 512, 128, ACT_GELU);
  __syncthreads();
  gemm_tile_g(sH, 516, Wt2, 512, bt2, sCat + 128, 644, 512, 512, ACT_NONE);
  __syncthreads();
  gemm_tile_g(sCat, 644, Wc1, 512, bc1, sA, 1028, 512, 640, ACT_GELU);
  __syncthreads();
  gemm_tile_g(sA, 1028, Wc2, 128, bc2, sH, 516, 128, 512, ACT_NONE);
  __syncthreads();
  for (int idx = threadIdx.x; idx < 16 * 128; idx += 256)
    cond[(size_t)(b0 + (idx >> 7)) * 128 + (idx & 127)] = sH[(idx >> 7) * 516 + (idx & 127)];
}

// ---------------------------------------------------------------- k_init
__global__ __launch_bounds__(128) void k_init(const int* atype, const float* semb,
                                              const float* gemb, float* h, float* hl) {
  int n = blockIdx.x, c = threadIdx.x;
  int z = atype[n];
  h[(size_t)n * 128 + c] = semb[(size_t)z * 128 + c];
  hl[(size_t)n * 128 + c] = gemb[(size_t)z * 128 + c];
}

// ---------------------------------------------------------------- k_edge_attr
__global__ __launch_bounds__(256) void k_edge_attr(
    const float* dist, const int* etype, const float* W1, const float* b1,
    const float* W2, const float* b2, const float* bemb, bf16_t* attr) {
  extern __shared__ char smc[];
  bf16_t* sW2 = (bf16_t*)smc;                 // 128*136*2 = 34816 (transposed)
  float* sHid = (float*)(smc + 34816);        // 16*132*4  = 8448
  int b = blockIdx.x;
  stage_w_t(sW2, W2, 128, 128);
  size_t ebase = (size_t)b * 1024;
  int wv = threadIdx.x >> 5, lane = threadIdx.x & 31;
  int n0 = wv * 16, nn = n0 + (lane & 15), mb = (lane >> 4) * 8;
  for (int mt = 0; mt < 64; ++mt) {
    __syncthreads();
    for (int idx = threadIdx.x; idx < 16 * 128; idx += 256) {
      int r = idx >> 7, c = idx & 127;
      float dv = dist[ebase + mt * 16 + r];
      sHid[r * 132 + c] = fmaxf(dv * W1[c] + b1[c], 0.f);
    }
    __syncthreads();
    v8f acc = {};
#pragma unroll
    for (int k0 = 0; k0 < 128; k0 += 32)
      acc = wmma_bf(load_a(sHid, 132, k0), load_b_t(sW2, 136, k0, n0), acc);
#pragma unroll
    for (int r = 0; r < 8; ++r) {
      size_t e = ebase + mt * 16 + mb + r;
      int et = etype[e];
      attr[e * 128 + nn] = (bf16_t)((acc[r] + b2[nn]) * bemb[(size_t)et * 128 + nn]);
    }
  }
}

// ---------------------------------------------------------------- k_gemm_x1
__global__ __launch_bounds__(256) void k_gemm_x1(const float* h, const float* lin1,
                                                 bf16_t* x1) {
  extern __shared__ char smc[];
  bf16_t* sW = (bf16_t*)smc;  // 128*136 bf16 transposed
  stage_w_t(sW, lin1, 128, 128);
  __syncthreads();
  int wv = threadIdx.x >> 5, lane = threadIdx.x & 31;
  int n0 = wv * 16, nn = n0 + (lane & 15), mb = (lane >> 4) * 8;
  for (int mt = 0; mt < 8; ++mt) {
    const float* Arow = h + ((size_t)blockIdx.x * 128 + mt * 16) * 128;
    v8f acc = {};
#pragma unroll
    for (int k0 = 0; k0 < 128; k0 += 32)
      acc = wmma_bf(load_a(Arow, 128, k0), load_b_t(sW, 136, k0, n0), acc);
#pragma unroll
    for (int r = 0; r < 8; ++r)
      x1[((size_t)blockIdx.x * 128 + mt * 16 + mb + r) * 128 + nn] = (bf16_t)acc[r];
  }
}

// ---------------------------------------------------------------- k_schnet
struct SchP {
  const float *F1, *f1b, *F2, *f2b, *linW, *linb, *lin1W, *lin2W, *lin2b;
};

#if __has_builtin(__builtin_amdgcn_tensor_load_to_lds)
#define HAVE_TDM 1
typedef __attribute__((ext_vector_type(4))) unsigned int v4u;
typedef __attribute__((ext_vector_type(8))) int v8i;
typedef __attribute__((ext_vector_type(4))) int v4i;
#endif

// LDS layout (dynamic): sWA@0 (34816) sWB@34816 (34816) sAgg@69632 (32*132*4)
// sTmp@86528 (16*132*4) sX1@94976 (32*128*2) sCm@103168 (4096) -> 107264
__global__ __launch_bounds__(256) void k_schnet(SchP p, const bf16_t* attr,
                                                const bf16_t* x1, const float* Cm,
                                                float* h) {
  extern __shared__ char smc[];
  bf16_t* sWA = (bf16_t*)smc;
  bf16_t* sWB = (bf16_t*)(smc + 34816);
  float* sAgg = (float*)(smc + 69632);
  float* sTmp = (float*)(smc + 86528);
  bf16_t* sX1 = (bf16_t*)(smc + 94976);
  float* sCmL = (float*)(smc + 103168);
  int b = blockIdx.x;
  stage_w_t(sWA, p.F1, 128, 128);
  stage_w_t(sWB, p.F2, 128, 128);
#if defined(HAVE_TDM)
  if (threadIdx.x < 32) {
    // TDM: 1-D tile of 4096 bf16 (x1 rows of this molecule) -> LDS @94976
    unsigned long long ga = (unsigned long long)(const void*)(x1 + (size_t)b * 4096);
    v4u g0;
    g0[0] = 1u;                                   // count=1 descriptor
    g0[1] = 94976u;                               // lds_addr
    g0[2] = (unsigned)(ga & 0xffffffffu);         // global_addr[31:0]
    g0[3] = (unsigned)((ga >> 32) & 0x01ffffffu) | (2u << 30);  // addr[56:32]|type=2
    v8i g1;
    g1[0] = (int)(1u << 16);     // data_size=1 (2 bytes), no multicast
    g1[1] = (int)(4096u << 16);  // tensor_dim0[15:0] @ bits63:48
    g1[2] = (int)(1u << 16);     // tensor_dim0 hi=0, tensor_dim1=1
    g1[3] = (int)(4096u << 16);  // tile_dim0 = 4096
    g1[4] = 1;                   // tile_dim1 = 1
    g1[5] = 4096;                // tensor_dim0_stride
    g1[6] = 0; g1[7] = 0;
    v4i z4 = {0, 0, 0, 0};
#if defined(__clang_major__) && __clang_major__ >= 23
    v8i z8 = {0, 0, 0, 0, 0, 0, 0, 0};
    __builtin_amdgcn_tensor_load_to_lds(g0, g1, z4, z4, z8, 0);
#else
    __builtin_amdgcn_tensor_load_to_lds(g0, g1, z4, z4, 0);
#endif
    __builtin_amdgcn_s_wait_tensorcnt(0);
  }
#else
  for (int idx = threadIdx.x; idx < 32 * 128; idx += 256)
    sX1[idx] = x1[(size_t)b * 4096 + idx];
#endif
  for (int idx = threadIdx.x; idx < 1024; idx += 256)
    sCmL[idx] = Cm[(size_t)b * 1024 + idx];
  for (int idx = threadIdx.x; idx < 32 * 132; idx += 256) sAgg[idx] = 0.f;
  __syncthreads();
  int wv = threadIdx.x >> 5, lane = threadIdx.x & 31;
  int n0 = wv * 16, nn = n0 + (lane & 15), mb = (lane >> 4) * 8;
  for (int mt = 0; mt < 64; ++mt) {
    size_t e0 = (size_t)b * 1024 + mt * 16;
    const bf16_t* Arow = attr + e0 * 128;
    if (mt < 63) __builtin_prefetch(attr + (e0 + 16) * 128, 0, 1);
    v8f acc = {};
#pragma unroll
    for (int k0 = 0; k0 < 128; k0 += 32)
      acc = wmma_bf(load_a(Arow, 128, k0), load_b_t(sWA, 136, k0, n0), acc);
#pragma unroll
    for (int r = 0; r < 8; ++r)
      sTmp[(mb + r) * 132 + nn] = apply_act(acc[r] + p.f1b[nn], ACT_SSP);
    __syncthreads();
    v8f acc2 = {};
#pragma unroll
    for (int k0 = 0; k0 < 128; k0 += 32)
      acc2 = wmma_bf(load_a(sTmp, 132, k0), load_b_t(sWB, 136, k0, n0), acc2);
    int iloc = mt >> 1, j0 = (mt & 1) * 16;
    float partial = 0.f;
#pragma unroll
    for (int r = 0; r < 8; ++r) {
      int j = j0 + mb + r;
      float wf = (acc2[r] + p.f2b[nn]) * sCmL[iloc * 32 + j];
      partial += wf * (float)sX1[j * 128 + nn];
    }
    // lanes l and l+16 hold the same column nn; waves own disjoint columns
    partial += __shfl_xor(partial, 16, 32);
    if (lane < 16) sAgg[iloc * 132 + nn] += partial;
    __syncthreads();
  }
  stage_w_t(sWA, p.lin2W, 128, 128);
  stage_w_t(sWB, p.linW, 128, 128);
  __syncthreads();
  for (int it = 0; it < 2; ++it) {
    const float* Aag = sAgg + it * 16 * 132;
    v8f acc = {};
#pragma unroll
    for (int k0 = 0; k0 < 128; k0 += 32)
      acc = wmma_bf(load_a(Aag, 132, k0), load_b_t(sWA, 136, k0, n0), acc);
#pragma unroll
    for (int r = 0; r < 8; ++r)
      sTmp[(mb + r) * 132 + nn] = apply_act(acc[r] + p.lin2b[nn], ACT_SSP);
    __syncthreads();
    v8f acc2 = {};
#pragma unroll
    for (int k0 = 0; k0 < 128; k0 += 32)
      acc2 = wmma_bf(load_a(sTmp, 132, k0), load_b_t(sWB, 136, k0, n0), acc2);
#pragma unroll
    for (int r = 0; r < 8; ++r) {
      size_t node = (size_t)b * 32 + it * 16 + mb + r;
      h[node * 128 + nn] += acc2[r] + p.linb[nn];
    }
    __syncthreads();
  }
}

// ---------------------------------------------------------------- k_gin
struct GinP { const float *Wa, *ba, *Wb, *bb; };

// LDS: sWA@0 sWB@34816 sHl@69632 (16384) sS@86016 (8448) sTmp@94464 (8448)
// sLm@102912 (4096) -> 107008
__global__ __launch_bounds__(256) void k_gin(GinP p, const bf16_t* attr,
                                             const int* etype, float* hl, int layer) {
  extern __shared__ char smc[];
  bf16_t* sWA = (bf16_t*)smc;
  bf16_t* sWB = (bf16_t*)(smc + 34816);
  float* sHl = (float*)(smc + 69632);
  float* sS = (float*)(smc + 86016);
  float* sTmp = (float*)(smc + 94464);
  float* sLm = (float*)(smc + 102912);
  int b = blockIdx.x;
  stage_w_t(sWA, p.Wa, 128, 128);
  stage_w_t(sWB, p.Wb, 128, 128);
  for (int idx = threadIdx.x; idx < 32 * 128; idx += 256)
    sHl[idx] = hl[(size_t)b * 4096 + idx];
  for (int idx = threadIdx.x; idx < 1024; idx += 256)
    sLm[idx] = (etype[(size_t)b * 1024 + idx] > 0) ? 1.f : 0.f;
  __syncthreads();
  int wv = threadIdx.x >> 5, lane = threadIdx.x & 31;
  int n0 = wv * 16, nn = n0 + (lane & 15), mb = (lane >> 4) * 8;
  for (int it = 0; it < 2; ++it) {
    int i0 = it * 16;
    for (int idx = threadIdx.x; idx < 16 * 128; idx += 256) {
      int r = idx >> 7, c = idx & 127;
      int ii = i0 + r;
      const bf16_t* arow = attr + ((size_t)b * 1024 + ii * 32) * 128 + c;
      float s = 0.f;
#pragma unroll 4
      for (int j = 0; j < 32; ++j) {
        float m = sHl[j * 128 + c] + (float)arow[(size_t)j * 128];
        s += fmaxf(m, 0.f) * sLm[ii * 32 + j];
      }
      sS[r * 132 + c] = sHl[ii * 128 + c] + s;
    }
    __syncthreads();
    v8f acc = {};
#pragma unroll
    for (int k0 = 0; k0 < 128; k0 += 32)
      acc = wmma_bf(load_a(sS, 132, k0), load_b_t(sWA, 136, k0, n0), acc);
#pragma unroll
    for (int r = 0; r < 8; ++r)
      sTmp[(mb + r) * 132 + nn] = fmaxf(acc[r] + p.ba[nn], 0.f);
    __syncthreads();
    v8f acc2 = {};
#pragma unroll
    for (int k0 = 0; k0 < 128; k0 += 32)
      acc2 = wmma_bf(load_a(sTmp, 132, k0), load_b_t(sWB, 136, k0, n0), acc2);
#pragma unroll
    for (int r = 0; r < 8; ++r) {
      int ii = i0 + mb + r;
      float out = acc2[r] + p.bb[nn];
      if (layer < 3) out = fmaxf(out, 0.f);
      hl[((size_t)b * 32 + ii) * 128 + nn] = out + sHl[ii * 128 + nn];
    }
    __syncthreads();
  }
}

// ---------------------------------------------------------------- k_addcond
__global__ __launch_bounds__(128) void k_addcond(const float* h, float* hl,
                                                 const float* cond, float* hg) {
  int node = blockIdx.x, c = threadIdx.x, b = node >> 5;
  float cv = cond[(size_t)b * 128 + c];
  hg[(size_t)node * 128 + c] = h[(size_t)node * 128 + c] + cv;
  hl[(size_t)node * 128 + c] += cv;
}

// ---------------------------------------------------------------- k_pair
struct PairP { const float *W1, *b1, *W2, *b2, *W3, *b3; };

// LDS: sW1t@0 (128*264*2=67584) sW2t@67584 (64*136*2... stored [64][264]? no:
// [N=64][K+8=136] -> 64*136*2=17408) sHf@84992 (16384) sA@101376 (16*260*4=16640)
// sH1@118016 (16*132*4=8448) sH2@126464 (4096) -> 130560
__global__ __launch_bounds__(256) void k_pair(PairP pg, PairP pl, const float* hg,
                                              const float* hlc, const bf16_t* attrG,
                                              const bf16_t* attrL, const float* validf,
                                              const float* localf, float* invG,
                                              float* invL) {
  extern __shared__ char smc[];
  bf16_t* sW1 = (bf16_t*)smc;
  bf16_t* sW2 = (bf16_t*)(smc + 67584);
  float* sHf = (float*)(smc + 84992);
  float* sA = (float*)(smc + 101376);
  float* sH1 = (float*)(smc + 118016);
  float* sH2 = (float*)(smc + 126464);
  int b = blockIdx.x, enc = blockIdx.y;
  const PairP p = enc ? pl : pg;
  const float* hf = enc ? hlc : hg;
  const bf16_t* attr = enc ? attrL : attrG;
  const float* mask = enc ? localf : validf;
  float* inv = enc ? invL : invG;
  stage_w_t(sW1, p.W1, 256, 128);
  stage_w_t(sW2, p.W2, 128, 64);
  for (int idx = threadIdx.x; idx < 32 * 128; idx += 256)
    sHf[idx] = hf[(size_t)b * 4096 + idx];
  __syncthreads();
  int wv = threadIdx.x >> 5, lane = threadIdx.x & 31;
  for (int mt = 0; mt < 64; ++mt) {
    int i = mt >> 1, j0 = (mt & 1) * 16;
    size_t e0 = (size_t)b * 1024 + i * 32 + j0;
    for (int idx = threadIdx.x; idx < 16 * 256; idx += 256) {
      int r = idx >> 8, c = idx & 255;
      sA[r * 260 + c] = (c < 128) ? sHf[i * 128 + c] * sHf[(j0 + r) * 128 + c]
                                  : (float)attr[(e0 + r) * 128 + (c - 128)];
    }
    __syncthreads();
    {
      int n0 = wv * 16, nn = n0 + (lane & 15), mb = (lane >> 4) * 8;
      v8f acc = {};
#pragma unroll
      for (int k0 = 0; k0 < 256; k0 += 32)
        acc = wmma_bf(load_a(sA, 260, k0), load_b_t(sW1, 264, k0, n0), acc);
#pragma unroll
      for (int r = 0; r < 8; ++r)
        sH1[(mb + r) * 132 + nn] = fmaxf(acc[r] + p.b1[nn], 0.f);
    }
    __syncthreads();
    if (wv < 4) {
      int n0 = wv * 16, nn = n0 + (lane & 15), mb = (lane >> 4) * 8;
      v8f acc = {};
#pragma unroll
      for (int k0 = 0; k0 < 128; k0 += 32)
        acc = wmma_bf(load_a(sH1, 132, k0), load_b_t(sW2, 136, k0, n0), acc);
#pragma unroll
      for (int r = 0; r < 8; ++r)
        sH2[(mb + r) * 64 + nn] = fmaxf(acc[r] + p.b2[nn], 0.f);
    }
    __syncthreads();
    if (threadIdx.x < 16) {
      int r = threadIdx.x;
      float dot = p.b3[0];
      for (int c = 0; c < 64; ++c) dot += sH2[r * 64 + c] * p.W3[c];
      inv[e0 + r] = dot * mask[e0 + r];
    }
    __syncthreads();
  }
}

// ---------------------------------------------------------------- launch
extern "C" void kernel_launch(void* const* d_in, const int* in_sizes, int n_in,
                              void* d_out, int out_size, void* d_ws, size_t ws_size,
                              hipStream_t stream) {
  (void)in_sizes; (void)n_in; (void)out_size; (void)ws_size;
  const float* pos = (const float*)d_in[0];
  const float* spectrum = (const float*)d_in[1];
  int pi = 2;
  auto F = [&]() { return (const float*)d_in[pi++]; };
  // params leaves: jax pytree order (sorted dict keys, lists in order)
  const float *Wc1 = F(), *bc1 = F(), *Wc2 = F(), *bc2 = F();                 // cond
  const float *embG = F(), *W1g = F(), *b1g = F(), *W2g = F(), *b2g = F();    // eenc_g
  const float *embL = F(), *W1l = F(), *b1l = F(), *W2l = F(), *b2l = F();    // eenc_l
  GinP gin[4];
  for (int l = 0; l < 4; ++l) { gin[l].Wa = F(); gin[l].ba = F(); gin[l].Wb = F(); gin[l].bb = F(); }
  const float* ginEmb = F();                                                  // gin_emb
  PairP pg, pl;
  pg.W1 = F(); pg.b1 = F(); pg.W2 = F(); pg.b2 = F(); pg.W3 = F(); pg.b3 = F();  // mlp_g
  pl.W1 = F(); pl.b1 = F(); pl.W2 = F(); pl.b2 = F(); pl.W3 = F(); pl.b3 = F();  // mlp_l
  SchP sch[6];  // schnet: keys sorted filt, lin, lin1, lin2
  for (int l = 0; l < 6; ++l) {
    sch[l].F1 = F(); sch[l].f1b = F(); sch[l].F2 = F(); sch[l].f2b = F();
    sch[l].linW = F(); sch[l].linb = F();
    sch[l].lin1W = F();
    sch[l].lin2W = F(); sch[l].lin2b = F();
  }
  const float* schEmb = F();                                                  // schnet_emb
  const float *Ws1 = F(), *bs1 = F(), *Ws2 = F(), *bs2 = F();                 // spec
  const float *Wt1 = F(), *bt1 = F(), *Wt2 = F(), *bt2 = F();                 // temb
  const int* atype = (const int*)d_in[pi++];
  const int* tstep = (const int*)d_in[pi++];
  const int* bond = (const int*)d_in[pi++];

  // workspace carve
  char* w = (char*)d_ws;
  auto alloc = [&](size_t bytes) { char* p = w; w += (bytes + 255) & ~(size_t)255; return p; };
  float* cond = (float*)alloc(128 * 128 * 4);
  float* dist = (float*)alloc((size_t)NE * 4);
  int* etype = (int*)alloc((size_t)NE * 4);
  float* Cm = (float*)alloc((size_t)NE * 4);
  float* validf = (float*)alloc((size_t)NE * 4);
  float* localf = (float*)alloc((size_t)NE * 4);
  float* h = (float*)alloc((size_t)4096 * 128 * 4);
  float* hl = (float*)alloc((size_t)4096 * 128 * 4);
  float* hg = (float*)alloc((size_t)4096 * 128 * 4);
  bf16_t* x1 = (bf16_t*)alloc((size_t)4096 * 128 * 2);
  bf16_t* attrG = (bf16_t*)alloc((size_t)NE * 128 * 2);
  bf16_t* attrL = (bf16_t*)alloc((size_t)NE * 128 * 2);

  // output carve (return order: inv_g, inv_l, edge_type, el, valid, local)
  float* out = (float*)d_out;
  float* invG = out;
  float* invL = out + NE;
  float* oEt = out + 2 * (size_t)NE;
  float* oEl = out + 3 * (size_t)NE;
  float* oValid = out + 4 * (size_t)NE;
  float* oLocal = out + 5 * (size_t)NE;

  k_graph<<<128, 1024, 0, stream>>>(pos, bond, dist, etype, Cm, validf, localf,
                                    oEt, oEl, oValid, oLocal);
  k_cond<<<8, 256, 140032, stream>>>(spectrum, tstep, Ws1, bs1, Ws2, bs2, Wt1,
                                     bt1, Wt2, bt2, Wc1, bc1, Wc2, bc2, cond);
  k_init<<<4096, 128, 0, stream>>>(atype, schEmb, ginEmb, h, hl);
  k_edge_attr<<<128, 256, 43264, stream>>>(dist, etype, W1g, b1g, W2g, b2g, embG, attrG);
  k_edge_attr<<<128, 256, 43264, stream>>>(dist, etype, W1l, b1l, W2l, b2l, embL, attrL);
  for (int l = 0; l < 6; ++l) {
    k_gemm_x1<<<32, 256, 34816, stream>>>(h, sch[l].lin1W, x1);
    k_schnet<<<128, 256, 107264, stream>>>(sch[l], attrG, x1, Cm, h);
  }
  for (int l = 0; l < 4; ++l)
    k_gin<<<128, 256, 107008, stream>>>(gin[l], attrL, etype, hl, l);
  k_addcond<<<4096, 128, 0, stream>>>(h, hl, cond, hg);
  k_pair<<<dim3(128, 2), 256, 130560, stream>>>(pg, pl, hg, hl, attrG, attrL,
                                                validf, localf, invG, invL);
}
